// WeatherAwareSTGATLayer_65249143161021
// MI455X (gfx1250) — matrix operand order repaired and checked by reading, not compiled, not run on hardware
//
#include <hip/hip_runtime.h>
#include <hip/hip_bf16.h>

// ---------------------------------------------------------------------------
// Problem dimensions (compile-time, from the reference)
// ---------------------------------------------------------------------------
#define B_   2
#define S_   64
#define N_   200
#define D_   256
#define WD_  32
#define H_   8
#define HD_  32
#define M_   (B_ * S_ * N_)   // 25600 token rows
#define M2_  (B_ * S_)        // 128
#define KFF_ (D_ + WD_)       // 288

// gfx1250 async global->LDS staging (ASYNCcnt path). Set to 0 to fall back to
// plain global_load_b128 + ds_store_b128 staging.
#define USE_ASYNC_LDS 1

typedef __bf16 bf16;
typedef __attribute__((ext_vector_type(16))) bf16  v16bf;
typedef __attribute__((ext_vector_type(8)))  float v8f;

struct __attribute__((aligned(16))) U128 { unsigned int x, y, z, w; };

// ---------------------------------------------------------------------------
// fp32 -> bf16 elementwise convert
// ---------------------------------------------------------------------------
__global__ __launch_bounds__(256) void cvt_bf16_kernel(const float* __restrict__ src,
                                                       bf16* __restrict__ dst, int n) {
  for (size_t i = (size_t)blockIdx.x * 256 + threadIdx.x; i < (size_t)n;
       i += (size_t)gridDim.x * 256)
    dst[i] = (bf16)src[i];
}

// ---------------------------------------------------------------------------
// Weight pack: W[K][Nc] fp32 (row-major) -> Wt[Nc][K] bf16 (transposed).
// ---------------------------------------------------------------------------
__global__ __launch_bounds__(256) void pack_wt_kernel(const float* __restrict__ w,
                                                      bf16* __restrict__ wt, int K, int Nc) {
  for (size_t i = (size_t)blockIdx.x * 256 + threadIdx.x; i < (size_t)K * Nc;
       i += (size_t)gridDim.x * 256) {
    int k = (int)(i / Nc), n = (int)(i % Nc);
    wt[(size_t)n * K + k] = (bf16)w[i];
  }
}

// ---------------------------------------------------------------------------
// WMMA bf16 GEMM:  C[M][Nn] (f32) = A[M][K] (bf16) @ Wt[Nn][K]^T (bf16) + bias
//
// Block tile 64(M) x 256(N); 8 waves as 2x4; wave tile 32x64 -> 2x4 = 8
// v_wmma_f32_16x16x32_bf16 per K=32 step (A-fragments reused over 4 B-frags).
// Double-buffered LDS stages, async global->LDS staging, one barrier/k-step.
// M % 64 == 0, K % 32 == 0, Nn % 256 == 0 guaranteed by problem sizes.
// ---------------------------------------------------------------------------
__global__ __launch_bounds__(256)
void gemm_wmma_bf16_kernel(const bf16* __restrict__ A, const bf16* __restrict__ Wt,
                           const float* __restrict__ bias, float* __restrict__ C,
                           int K, int Nn, int relu) {
  __shared__ bf16 ldsA[2][64 * 40];    // per stage: 64 rows x 32 halves (+pad)
  __shared__ bf16 ldsB[2][256 * 40];   // per stage: 256 cols x 32 halves (+pad)

  const int n0   = blockIdx.x * 256;
  const int m0   = blockIdx.y * 64;
  const int t    = threadIdx.x;
  const int lane = t & 31;
  const int wave = t >> 5;
  const int waveM = wave & 1;       // 2 waves across M (32 rows each)
  const int waveN = wave >> 1;      // 4 waves across N (64 cols each)
  const int laneHi = (lane >> 4) & 1;
  const int lane15 = lane & 15;

  // Per-thread staging addresses (A: 16B/thread, B: 64B/thread)
  const int aRow = t >> 2, aSeg = t & 3;
  const bf16* gA = A + (size_t)(m0 + aRow) * K + aSeg * 8;
  const bf16* gB = Wt + (size_t)(n0 + t) * K;
  bf16* lA0 = &ldsA[0][aRow * 40 + aSeg * 8];
  bf16* lA1 = &ldsA[1][aRow * 40 + aSeg * 8];
  bf16* lB0 = &ldsB[0][t * 40];
  bf16* lB1 = &ldsB[1][t * 40];

  v8f acc[2][4];
#pragma unroll
  for (int a = 0; a < 2; ++a)
#pragma unroll
    for (int b = 0; b < 4; ++b)
#pragma unroll
      for (int r = 0; r < 8; ++r) acc[a][b][r] = 0.0f;

  const int kTiles = K >> 5;

  // ---- tile loader: stage global tile kt into LDS stage sel
#if USE_ASYNC_LDS
  #define LOAD_TILE(kt, lA, lB)                                                    \
    do {                                                                           \
      const bf16* ga = gA + ((kt) << 5);                                           \
      asm volatile("global_load_async_to_lds_b128 %0, %1, off"                     \
                   :: "v"((unsigned int)(unsigned long long)(lA)), "v"(ga)         \
                   : "memory");                                                    \
      const bf16* gb = gB + ((kt) << 5);                                           \
      _Pragma("unroll")                                                            \
      for (int sg = 0; sg < 4; ++sg) {                                             \
        asm volatile("global_load_async_to_lds_b128 %0, %1, off"                   \
                     :: "v"((unsigned int)(unsigned long long)((lB) + sg * 8)),    \
                        "v"(gb + sg * 8)                                           \
                     : "memory");                                                  \
      }                                                                            \
    } while (0)
  #define DRAIN_TILES() asm volatile("s_wait_asynccnt 0x0" ::: "memory")
#else
  #define LOAD_TILE(kt, lA, lB)                                                    \
    do {                                                                           \
      const U128 va = *(const U128*)(gA + ((kt) << 5));                            \
      *(U128*)(lA) = va;                                                           \
      const U128* sb = (const U128*)(gB + ((kt) << 5));                            \
      U128* db = (U128*)(lB);                                                      \
      _Pragma("unroll")                                                            \
      for (int sg = 0; sg < 4; ++sg) db[sg] = sb[sg];                              \
    } while (0)
  #define DRAIN_TILES() ((void)0)
#endif

  LOAD_TILE(0, lA0, lB0);
  DRAIN_TILES();
  __syncthreads();

  for (int kt = 0; kt < kTiles; ++kt) {
    const int stage = kt & 1;
    if (kt + 1 < kTiles) {          // stage next tile into the other buffer
      if (stage) LOAD_TILE(kt + 1, lA0, lB0);
      else       LOAD_TILE(kt + 1, lA1, lB1);
    }
    if (kt + 2 < kTiles)            // gfx1250 global_prefetch_b8 for k+2
      __builtin_prefetch(gA + ((kt + 2) << 5), 0, 1);

    const bf16* sA = ldsA[stage];
    const bf16* sB = ldsB[stage];

    // A fragments (16x32 bf16): per ISA 16-bit A layout, lanes 0-15 hold
    // K {0..7,16..23}, lanes 16-31 hold K {8..15,24..31}; row = M index.
    v16bf aF[2], bF[4];
#pragma unroll
    for (int tm = 0; tm < 2; ++tm) {
      const int row = waveM * 32 + tm * 16 + lane15;
      U128 x0 = *(const U128*)(&sA[row * 40 + laneHi * 8]);
      U128 x1 = *(const U128*)(&sA[row * 40 + 16 + laneHi * 8]);
      const bf16* h0 = (const bf16*)&x0;
      const bf16* h1 = (const bf16*)&x1;
#pragma unroll
      for (int j = 0; j < 8; ++j) { aF[tm][j] = h0[j]; aF[tm][8 + j] = h1[j]; }
    }
    // B fragments (32x16 bf16): col = N index (lane%16); lanes 0-15 hold
    // K {0..7,16..23}, lanes 16-31 hold K {8..15,24..31} (same as A).
#pragma unroll
    for (int tn = 0; tn < 4; ++tn) {
      const int col = waveN * 64 + tn * 16 + lane15;
      U128 y0 = *(const U128*)(&sB[col * 40 + laneHi * 8]);
      U128 y1 = *(const U128*)(&sB[col * 40 + 16 + laneHi * 8]);
      const bf16* h0 = (const bf16*)&y0;
      const bf16* h1 = (const bf16*)&y1;
#pragma unroll
      for (int j = 0; j < 8; ++j) { bF[tn][j] = h0[j]; bF[tn][8 + j] = h1[j]; }
    }

#pragma unroll
    for (int tm = 0; tm < 2; ++tm)
#pragma unroll
      for (int tn = 0; tn < 4; ++tn)
        acc[tm][tn] = __builtin_amdgcn_wmma_f32_16x16x32_bf16(
            false, aF[tm], false, bF[tn], (short)0, acc[tm][tn], false, false);

    DRAIN_TILES();        // next stage fully resident in LDS
    __syncthreads();      // readers of this stage done; next stage published
  }

  // Store: D layout lane L / vgpr r -> row = r + 8*(L>=16), col = L%16.
#pragma unroll
  for (int tm = 0; tm < 2; ++tm) {
#pragma unroll
    for (int tn = 0; tn < 4; ++tn) {
      const int col = n0 + waveN * 64 + tn * 16 + lane15;
      const float bc = bias ? bias[col] : 0.0f;
#pragma unroll
      for (int r = 0; r < 8; ++r) {
        const int row = m0 + waveM * 32 + tm * 16 + laneHi * 8 + r;
        float v = acc[tm][tn][r] + bc;
        if (relu) v = fmaxf(v, 0.0f);
        C[(size_t)row * Nn + col] = v;
      }
    }
  }
#undef LOAD_TILE
#undef DRAIN_TILES
}

// ---------------------------------------------------------------------------
// Spatial masked attention. One block per (b,s,h); K/V tiles (200x32) in LDS.
// ---------------------------------------------------------------------------
__global__ __launch_bounds__(256)
void spatial_attn_kernel(const float* __restrict__ Q, const float* __restrict__ Km,
                         const float* __restrict__ V, const float* __restrict__ adj,
                         float* __restrict__ av) {
  __shared__ float Kl[N_ * 33];
  __shared__ float Vl[N_ * 33];
  const int blk  = blockIdx.x;          // b*S*H + s*H + h
  const int bs   = blk / H_;
  const int h    = blk % H_;
  const int b    = bs / S_;
  const int t    = threadIdx.x;
  const int lane = t & 31;
  const int wave = t >> 5;

  for (int idx = t; idx < N_ * HD_; idx += 256) {
    const int n = idx >> 5, d = idx & 31;
    const size_t g = ((size_t)(bs * N_ + n)) * D_ + h * HD_ + d;
    Kl[n * 33 + d] = Km[g];
    Vl[n * 33 + d] = V[g];
  }
  __syncthreads();

  const float invScale = 0.17677669529663687f;   // 1/sqrt(32)
  for (int r = wave; r < N_; r += 8) {
    const float qown = Q[((size_t)(bs * N_ + r)) * D_ + h * HD_ + lane];
    float sc[7];
#pragma unroll
    for (int j = 0; j < 7; ++j) sc[j] = 0.0f;
#pragma unroll
    for (int d = 0; d < HD_; ++d) {
      const float qd = __shfl(qown, d, 32);
#pragma unroll
      for (int j = 0; j < 7; ++j) {
        const int m = lane + 32 * j;
        if (m < N_) sc[j] += qd * Kl[m * 33 + d];
      }
    }
    float mx = -1e30f;
#pragma unroll
    for (int j = 0; j < 7; ++j) {
      const int m = lane + 32 * j;
      if (m < N_) {
        const float a = adj[(size_t)b * N_ * N_ + (size_t)r * N_ + m];
        sc[j] = (a == 0.0f) ? -1e9f : sc[j] * invScale;
        mx = fmaxf(mx, sc[j]);
      } else sc[j] = -1e30f;
    }
#pragma unroll
    for (int o = 16; o > 0; o >>= 1) mx = fmaxf(mx, __shfl_xor(mx, o, 32));
    float p[7], sum = 0.0f;
#pragma unroll
    for (int j = 0; j < 7; ++j) {
      const int m = lane + 32 * j;
      p[j] = (m < N_) ? __expf(sc[j] - mx) : 0.0f;
      sum += p[j];
    }
#pragma unroll
    for (int o = 16; o > 0; o >>= 1) sum += __shfl_xor(sum, o, 32);
    const float inv = 1.0f / sum;

    float outd = 0.0f;    // lane = head-dim d
#pragma unroll
    for (int j = 0; j < 7; ++j) {
      const int lim = (j == 6) ? (N_ - 192) : 32;
      for (int i = 0; i < lim; ++i) {
        const float pm = __shfl(p[j], i, 32);
        outd += pm * Vl[(j * 32 + i) * 33 + lane];
      }
    }
    av[((size_t)(bs * N_ + r)) * D_ + h * HD_ + lane] = outd * inv;
  }
}

// ---------------------------------------------------------------------------
// Multi-scale temporal attention. One block per (b,n,h). S=64, scales {1,2,4}.
// ---------------------------------------------------------------------------
__global__ __launch_bounds__(256)
void temporal_attn_kernel(const float* __restrict__ Qt, const float* __restrict__ Kt,
                          const float* __restrict__ Vt, float* __restrict__ acc) {
  __shared__ float Ql[S_ * 33], Kl[S_ * 33], Vl[S_ * 33];
  __shared__ float Ks[S_ * 33], Vs[S_ * 33];
  const int blk  = blockIdx.x;
  const int b    = blk / (N_ * H_);
  const int n    = (blk / H_) % N_;
  const int h    = blk % H_;
  const int t    = threadIdx.x;
  const int lane = t & 31;
  const int wave = t >> 5;

  for (int idx = t; idx < S_ * HD_; idx += 256) {
    const int s = idx >> 5, d = idx & 31;
    const size_t g = ((size_t)((b * S_ + s) * N_ + n)) * D_ + h * HD_ + d;
    Ql[s * 33 + d] = Qt[g];
    Kl[s * 33 + d] = Kt[g];
    Vl[s * 33 + d] = Vt[g];
  }
  __syncthreads();

  float accd[8];
#pragma unroll
  for (int i = 0; i < 8; ++i) accd[i] = 0.0f;
  const float invScale = 0.17677669529663687f;

#pragma unroll
  for (int si = 0; si < 3; ++si) {
    const int sc_ = 1 << si;          // 1, 2, 4
    const int T   = S_ / sc_;         // 64, 32, 16
    for (int idx = t; idx < T * HD_; idx += 256) {   // averaged K/V
      const int tt = idx >> 5, d = idx & 31;
      float ks = 0.0f, vs = 0.0f;
      for (int j = 0; j < sc_; ++j) {
        ks += Kl[(tt * sc_ + j) * 33 + d];
        vs += Vl[(tt * sc_ + j) * 33 + d];
      }
      const float invs = 1.0f / (float)sc_;
      Ks[tt * 33 + d] = ks * invs;
      Vs[tt * 33 + d] = vs * invs;
    }
    __syncthreads();

    for (int i = 0; i < 8; ++i) {
      const int r = wave * 8 + i;
      const float qown = Ql[r * 33 + lane];
      float s0 = 0.0f, s1 = 0.0f;
#pragma unroll
      for (int d = 0; d < HD_; ++d) {
        const float qd = __shfl(qown, d, 32);
        s0 += qd * Ks[lane * 33 + d];
        if (T == 64) s1 += qd * Ks[(lane + 32) * 33 + d];
      }
      s0 = (lane < T) ? s0 * invScale : -1e30f;
      s1 = (T == 64) ? s1 * invScale : -1e30f;
      float mx = fmaxf(s0, s1);
#pragma unroll
      for (int o = 16; o > 0; o >>= 1) mx = fmaxf(mx, __shfl_xor(mx, o, 32));
      const float p0 = (lane < T) ? __expf(s0 - mx) : 0.0f;
      const float p1 = (T == 64) ? __expf(s1 - mx) : 0.0f;
      float sum = p0 + p1;
#pragma unroll
      for (int o = 16; o > 0; o >>= 1) sum += __shfl_xor(sum, o, 32);
      const float inv = 1.0f / sum;

      float outd = 0.0f;              // lane = head-dim d
      for (int tt = 0; tt < T; ++tt) {
        const float pv = (tt < 32) ? __shfl(p0, tt & 31, 32) : __shfl(p1, tt & 31, 32);
        outd += pv * Vs[tt * 33 + lane];
      }
      accd[i] += outd * inv;
    }
    __syncthreads();
  }

  const float third = 1.0f / 3.0f;
  for (int i = 0; i < 8; ++i) {
    const int r = wave * 8 + i;
    acc[((size_t)((b * S_ + r) * N_ + n)) * D_ + h * HD_ + lane] = accd[i] * third;
  }
}

// ---------------------------------------------------------------------------
// out = LayerNorm(resid + lin) * g + b   (one wave per 256-wide row)
// ---------------------------------------------------------------------------
__global__ __launch_bounds__(256)
void add_ln_kernel(const float* __restrict__ resid, const float* __restrict__ lin,
                   const float* __restrict__ g, const float* __restrict__ bta,
                   float* __restrict__ out) {
  const int t = threadIdx.x, lane = t & 31, wave = t >> 5;
  const int row = blockIdx.x * 8 + wave;
  const size_t base = (size_t)row * D_;
  float x[8], s = 0.0f, sq = 0.0f;
#pragma unroll
  for (int j = 0; j < 8; ++j) {
    const int d = j * 32 + lane;
    x[j] = resid[base + d] + lin[base + d];
    s += x[j];
    sq += x[j] * x[j];
  }
#pragma unroll
  for (int o = 16; o > 0; o >>= 1) {
    s += __shfl_xor(s, o, 32);
    sq += __shfl_xor(sq, o, 32);
  }
  const float mean = s * (1.0f / D_);
  const float var  = sq * (1.0f / D_) - mean * mean;
  const float rstd = rsqrtf(var + 1e-5f);
#pragma unroll
  for (int j = 0; j < 8; ++j) {
    const int d = j * 32 + lane;
    out[base + d] = (x[j] - mean) * rstd * g[d] + bta[d];
  }
}

// ---------------------------------------------------------------------------
// Weather projection: out[bs][d] = weather[bs] . wp_w[:,d] + wp_b[d]  (K=32)
// ---------------------------------------------------------------------------
__global__ __launch_bounds__(256)
void weather_proj_kernel(const float* __restrict__ we, const float* __restrict__ w,
                         const float* __restrict__ b, float* __restrict__ out) {
  __shared__ float wl[WD_];
  const int bs = blockIdx.x;
  if (threadIdx.x < WD_) wl[threadIdx.x] = we[bs * WD_ + threadIdx.x];
  __syncthreads();
  const int d = threadIdx.x;
  float s = b[d];
#pragma unroll
  for (int k = 0; k < WD_; ++k) s += wl[k] * w[k * D_ + d];
  out[bs * D_ + d] = s;
}

// ---------------------------------------------------------------------------
// hb = bf16(sp + broadcast(wproj))
// ---------------------------------------------------------------------------
__global__ __launch_bounds__(256)
void add_weather_kernel(const float* __restrict__ sp, const float* __restrict__ wproj,
                        bf16* __restrict__ hb) {
  for (size_t i = (size_t)blockIdx.x * 256 + threadIdx.x; i < (size_t)M_ * D_;
       i += (size_t)gridDim.x * 256) {
    const size_t m = i / D_;
    const int c = (int)(i % D_);
    const size_t bs = m / N_;
    hb[i] = (bf16)(sp[i] + wproj[bs * D_ + c]);
  }
}

// ---------------------------------------------------------------------------
// fused[m][0:256] = bf16(tp[m]);  fused[m][256:288] = bf16(weather[bs])
// ---------------------------------------------------------------------------
__global__ __launch_bounds__(256)
void build_fused_kernel(const float* __restrict__ tp, const float* __restrict__ we,
                        bf16* __restrict__ fused) {
  for (size_t i = (size_t)blockIdx.x * 256 + threadIdx.x; i < (size_t)M_ * KFF_;
       i += (size_t)gridDim.x * 256) {
    const size_t m = i / KFF_;
    const int c = (int)(i % KFF_);
    const float v = (c < D_) ? tp[m * D_ + c] : we[(m / N_) * WD_ + (c - D_)];
    fused[i] = (bf16)v;
  }
}

// ---------------------------------------------------------------------------
// Host-side orchestration
// ---------------------------------------------------------------------------
extern "C" void kernel_launch(void* const* d_in, const int* in_sizes, int n_in,
                              void* d_out, int out_size, void* d_ws, size_t ws_size,
                              hipStream_t stream) {
  (void)in_sizes; (void)n_in; (void)out_size; (void)ws_size;

  const float* node    = (const float*)d_in[0];
  const float* weather = (const float*)d_in[1];
  const float* adj     = (const float*)d_in[2];
  const float* sq_w = (const float*)d_in[3],  *sq_b = (const float*)d_in[4];
  const float* sk_w = (const float*)d_in[5],  *sk_b = (const float*)d_in[6];
  const float* sv_w = (const float*)d_in[7],  *sv_b = (const float*)d_in[8];
  const float* so_w = (const float*)d_in[9],  *so_b = (const float*)d_in[10];
  const float* sln_g = (const float*)d_in[11], *sln_b = (const float*)d_in[12];
  const float* wp_w = (const float*)d_in[13], *wp_b = (const float*)d_in[14];
  const float* tq_w = (const float*)d_in[15], *tq_b = (const float*)d_in[16];
  const float* tk_w = (const float*)d_in[17], *tk_b = (const float*)d_in[18];
  const float* tv_w = (const float*)d_in[19], *tv_b = (const float*)d_in[20];
  const float* to_w = (const float*)d_in[21], *to_b = (const float*)d_in[22];
  const float* tln_g = (const float*)d_in[23], *tln_b = (const float*)d_in[24];
  const float* ff_w = (const float*)d_in[25], *ff_b = (const float*)d_in[26];
  const float* fln_g = (const float*)d_in[27], *fln_b = (const float*)d_in[28];
  float* out = (float*)d_out;

  // Bump allocator over workspace (deterministic layout every call).
  char* base = (char*)d_ws;
  size_t off = 0;
  auto take = [&](size_t bytes) -> char* {
    char* p = base + off;
    off += (bytes + 255) & ~(size_t)255;
    return p;
  };
  const size_t WDD = (size_t)D_ * D_ * sizeof(bf16);
  const size_t MD2 = (size_t)M_ * D_ * sizeof(bf16);
  const size_t MD4 = (size_t)M_ * D_ * sizeof(float);

  bf16* Wq  = (bf16*)take(WDD);
  bf16* Wk  = (bf16*)take(WDD);
  bf16* Wv  = (bf16*)take(WDD);
  bf16* Wo  = (bf16*)take(WDD);
  bf16* Wtq = (bf16*)take(WDD);
  bf16* Wtk = (bf16*)take(WDD);
  bf16* Wtv = (bf16*)take(WDD);
  bf16* Wto = (bf16*)take(WDD);
  bf16* Wff = (bf16*)take((size_t)KFF_ * D_ * sizeof(bf16));
  bf16* XB  = (bf16*)take(MD2);   // reused: xb -> avb -> hb -> accb
  bf16* FUS = (bf16*)take((size_t)M_ * KFF_ * sizeof(bf16));
  float* S0 = (float*)take(MD4);
  float* S1 = (float*)take(MD4);
  float* S2 = (float*)take(MD4);
  float* S3 = (float*)take(MD4);
  float* S4 = (float*)take(MD4);
  float* S5 = (float*)take(MD4);
  float* WPJ = (float*)take((size_t)M2_ * D_ * sizeof(float));

  const dim3 blk256(256);
  const dim3 gemmGrid(D_ / 256, M_ / 64);     // (1, 400)

  // --- prep: pack weights (transposed bf16), convert activations
  hipLaunchKernelGGL(pack_wt_kernel, dim3(256), blk256, 0, stream, sq_w, Wq,  D_,  D_);
  hipLaunchKernelGGL(pack_wt_kernel, dim3(256), blk256, 0, stream, sk_w, Wk,  D_,  D_);
  hipLaunchKernelGGL(pack_wt_kernel, dim3(256), blk256, 0, stream, sv_w, Wv,  D_,  D_);
  hipLaunchKernelGGL(pack_wt_kernel, dim3(256), blk256, 0, stream, so_w, Wo,  D_,  D_);
  hipLaunchKernelGGL(pack_wt_kernel, dim3(256), blk256, 0, stream, tq_w, Wtq, D_,  D_);
  hipLaunchKernelGGL(pack_wt_kernel, dim3(256), blk256, 0, stream, tk_w, Wtk, D_,  D_);
  hipLaunchKernelGGL(pack_wt_kernel, dim3(256), blk256, 0, stream, tv_w, Wtv, D_,  D_);
  hipLaunchKernelGGL(pack_wt_kernel, dim3(256), blk256, 0, stream, to_w, Wto, D_,  D_);
  hipLaunchKernelGGL(pack_wt_kernel, dim3(256), blk256, 0, stream, ff_w, Wff, KFF_, D_);
  hipLaunchKernelGGL(cvt_bf16_kernel, dim3(2048), blk256, 0, stream, node, XB, M_ * D_);

  // --- spatial attention block
  hipLaunchKernelGGL(gemm_wmma_bf16_kernel, gemmGrid, blk256, 0, stream, XB, Wq, sq_b, S0, D_, D_, 0);
  hipLaunchKernelGGL(gemm_wmma_bf16_kernel, gemmGrid, blk256, 0, stream, XB, Wk, sk_b, S1, D_, D_, 0);
  hipLaunchKernelGGL(gemm_wmma_bf16_kernel, gemmGrid, blk256, 0, stream, XB, Wv, sv_b, S2, D_, D_, 0);
  hipLaunchKernelGGL(spatial_attn_kernel, dim3(B_ * S_ * H_), blk256, 0, stream, S0, S1, S2, adj, S3);
  hipLaunchKernelGGL(cvt_bf16_kernel, dim3(2048), blk256, 0, stream, S3, XB, M_ * D_);
  hipLaunchKernelGGL(gemm_wmma_bf16_kernel, gemmGrid, blk256, 0, stream, XB, Wo, so_b, S4, D_, D_, 0);
  hipLaunchKernelGGL(add_ln_kernel, dim3(M_ / 8), blk256, 0, stream, node, S4, sln_g, sln_b, S5); // sp

  // --- weather injection
  hipLaunchKernelGGL(weather_proj_kernel, dim3(M2_), blk256, 0, stream, weather, wp_w, wp_b, WPJ);
  hipLaunchKernelGGL(add_weather_kernel, dim3(2048), blk256, 0, stream, S5, WPJ, XB); // hb

  // --- temporal attention block
  hipLaunchKernelGGL(gemm_wmma_bf16_kernel, gemmGrid, blk256, 0, stream, XB, Wtq, tq_b, S1, D_, D_, 0);
  hipLaunchKernelGGL(gemm_wmma_bf16_kernel, gemmGrid, blk256, 0, stream, XB, Wtk, tk_b, S2, D_, D_, 0);
  hipLaunchKernelGGL(gemm_wmma_bf16_kernel, gemmGrid, blk256, 0, stream, XB, Wtv, tv_b, S3, D_, D_, 0);
  hipLaunchKernelGGL(temporal_attn_kernel, dim3(B_ * N_ * H_), blk256, 0, stream, S1, S2, S3, S4);
  hipLaunchKernelGGL(cvt_bf16_kernel, dim3(2048), blk256, 0, stream, S4, XB, M_ * D_);
  hipLaunchKernelGGL(gemm_wmma_bf16_kernel, gemmGrid, blk256, 0, stream, XB, Wto, to_b, S0, D_, D_, 0);
  hipLaunchKernelGGL(add_ln_kernel, dim3(M_ / 8), blk256, 0, stream, S5, S0, tln_g, tln_b, S1); // tp

  // --- feed-forward fusion with weather + final LN
  hipLaunchKernelGGL(build_fused_kernel, dim3(2048), blk256, 0, stream, S1, weather, FUS);
  hipLaunchKernelGGL(gemm_wmma_bf16_kernel, gemmGrid, blk256, 0, stream, FUS, Wff, ff_b, S2, KFF_, D_, 1);
  hipLaunchKernelGGL(add_ln_kernel, dim3(M_ / 8), blk256, 0, stream, node, S2, fln_g, fln_b, out);
}
